// EdgeDetourHead_62732292325617
// MI455X (gfx1250) — matrix operand order repaired and checked by reading, not compiled, not run on hardware
//
#include <hip/hip_runtime.h>

typedef __bf16 bf16;
typedef __attribute__((ext_vector_type(16))) __bf16 v16bf;
typedef __attribute__((ext_vector_type(8)))  __bf16 v8bf;
typedef __attribute__((ext_vector_type(8)))  float  v8f;

#define B_  2
#define N_  512
#define E_  128
#define H_  256
#define K1  384   // concat features without the euclid scalar
#define KS1 12    // layer-1 k-steps of 32
#define KS2 8     // layer-2 k-steps of 32
#define MT  32    // edges (M rows) per workgroup: two 16-row WMMA sub-tiles

// ---------------------------------------------------------------------------
// Pack W1 (385x256) rows 0..383 and W2 (256x256) into WMMA B-fragment layout:
// fragment (kstep,ntile): lane l, element e  ->  K = kstep*32 + (e>>3)*16 + (l>>4)*8 + (e&7),
//                                                N = ntile*16 + (l&15)
// so the main kernel loads one contiguous v16bf (32B) per lane per fragment.
// ---------------------------------------------------------------------------
__global__ void pack_weights(const float* __restrict__ W1,
                             const float* __restrict__ W2,
                             bf16* __restrict__ w1p, bf16* __restrict__ w2p) {
  int idx = blockIdx.x * blockDim.x + threadIdx.x;
  const int n1 = KS1 * 16 * 32 * 16;   // 98304
  const int n2 = KS2 * 16 * 32 * 16;   // 65536
  if (idx < n1) {
    int e = idx & 15, l = (idx >> 4) & 31, frag = idx >> 9;
    int kstep = frag >> 4, ntile = frag & 15;
    int K = kstep * 32 + ((e >> 3) * 16 + (l >> 4) * 8 + (e & 7));
    int Ncol = ntile * 16 + (l & 15);
    w1p[idx] = (bf16)W1[K * H_ + Ncol];
  } else if (idx < n1 + n2) {
    int j = idx - n1;
    int e = j & 15, l = (j >> 4) & 31, frag = j >> 9;
    int kstep = frag >> 4, ntile = frag & 15;
    int K = kstep * 32 + ((e >> 3) * 16 + (l >> 4) * 8 + (e & 7));
    int Ncol = ntile * 16 + (l & 15);
    w2p[j] = (bf16)W2[K * H_ + Ncol];
  }
}

// ---------------------------------------------------------------------------
// One workgroup = 8 waves, handles edges (b, i, j0..j0+31): two 16-row
// M sub-tiles so every B fragment fetched from L2 feeds 2 WMMAs.
// Wave w owns hidden tiles 2w and 2w+1 (16 columns each) for both layers.
// ---------------------------------------------------------------------------
__global__ __launch_bounds__(256) void mlp_edge(
    const float* __restrict__ node, const float* __restrict__ euclid,
    const bf16* __restrict__ w1p, const bf16* __restrict__ w2p,
    const float* __restrict__ W1,   // f32, for row 384 (euclid feature)
    const float* __restrict__ b1, const float* __restrict__ b2,
    const float* __restrict__ W3, const float* __restrict__ b3,
    float* __restrict__ predws) {
  __shared__ __align__(16) bf16 xt[MT * K1];   // 24 KB feature tile
  __shared__ __align__(16) bf16 h1[MT * H_];   // 16 KB hidden tile
  __shared__ float eu[MT];
  __shared__ float pl[MT];

  const int tid = threadIdx.x;
  const int blk = blockIdx.x;
  const int jt  = blk & 15;                 // N/32 = 16 tiles per row
  const int i   = (blk >> 4) & (N_ - 1);
  const int b   = blk >> 13;                // 16*512 = 2^13 tiles per batch
  const int j0  = jt * MT;

  // ---- build 32x384 bf16 feature tile: [hi | hj | |hi-hj|] ----
  const float* hi = node + ((size_t)b * N_ + i) * E_;
  for (int idx = tid; idx < MT * K1; idx += 256) {
    int m = idx / K1, k = idx - m * K1;
    int j = j0 + m;
    float v;
    if (k < E_) {
      v = hi[k];
    } else if (k < 2 * E_) {
      v = node[((size_t)b * N_ + j) * E_ + (k - E_)];
    } else {
      float a = hi[k - 2 * E_];
      float c = node[((size_t)b * N_ + j) * E_ + (k - 2 * E_)];
      v = fabsf(a - c);
    }
    xt[m * K1 + k] = (bf16)v;
  }
  if (tid < MT) {
    eu[tid] = euclid[((size_t)b * N_ + i) * N_ + j0 + tid];
    pl[tid] = 0.f;
  }
  __syncthreads();

  const int lane = tid & 31;
  const int wave = tid >> 5;
  const int m    = lane & 15;   // A-fragment row within sub-tile
  const int hh   = lane >> 4;   // half-wave selects K sub-range
  const int ncol = lane & 15;   // D/B column within n-tile
  const int nt0  = wave * 2, nt1 = wave * 2 + 1;

  // ---- layer 1: 32x384 @ 384x256 ----
  v8f a00 = {}, a01 = {}, a10 = {}, a11 = {};   // acc[mtile][ntile]
  for (int ks = 0; ks < KS1; ++ks) {
    union { v16bf v; v8bf h[2]; } A0, A1;
    const bf16* ab0 = &xt[m * K1 + ks * 32 + 8 * hh];
    const bf16* ab1 = ab0 + 16 * K1;
    A0.h[0] = *(const v8bf*)(ab0);        // K = 32ks + 8hh + [0..7]
    A0.h[1] = *(const v8bf*)(ab0 + 16);   // K = 32ks + 16 + 8hh + [0..7]
    A1.h[0] = *(const v8bf*)(ab1);
    A1.h[1] = *(const v8bf*)(ab1 + 16);
    v16bf B0 = *(const v16bf*)(w1p + (((ks * 16 + nt0) * 32 + lane) << 4));
    a00 = __builtin_amdgcn_wmma_f32_16x16x32_bf16(false, A0.v, false, B0,
                                                  (short)0, a00, false, false);
    a10 = __builtin_amdgcn_wmma_f32_16x16x32_bf16(false, A1.v, false, B0,
                                                  (short)0, a10, false, false);
    v16bf B1 = *(const v16bf*)(w1p + (((ks * 16 + nt1) * 32 + lane) << 4));
    a01 = __builtin_amdgcn_wmma_f32_16x16x32_bf16(false, A0.v, false, B1,
                                                  (short)0, a01, false, false);
    a11 = __builtin_amdgcn_wmma_f32_16x16x32_bf16(false, A1.v, false, B1,
                                                  (short)0, a11, false, false);
  }

  // euclid rank-1 term + bias + relu, write h1 tile (bf16) following D layout
  {
    float wl0 = W1[(size_t)K1 * H_ + nt0 * 16 + ncol];
    float wl1 = W1[(size_t)K1 * H_ + nt1 * 16 + ncol];
    float bb0 = b1[nt0 * 16 + ncol];
    float bb1 = b1[nt1 * 16 + ncol];
#pragma unroll
    for (int r = 0; r < 8; ++r) {
      int mm = r + 8 * hh;                      // D row within sub-tile
      float e0 = eu[mm], e1 = eu[16 + mm];
      float v00 = a00[r] + e0 * wl0 + bb0; v00 = v00 > 0.f ? v00 : 0.f;
      float v01 = a01[r] + e0 * wl1 + bb1; v01 = v01 > 0.f ? v01 : 0.f;
      float v10 = a10[r] + e1 * wl0 + bb0; v10 = v10 > 0.f ? v10 : 0.f;
      float v11 = a11[r] + e1 * wl1 + bb1; v11 = v11 > 0.f ? v11 : 0.f;
      h1[mm * H_ + nt0 * 16 + ncol]        = (bf16)v00;
      h1[mm * H_ + nt1 * 16 + ncol]        = (bf16)v01;
      h1[(16 + mm) * H_ + nt0 * 16 + ncol] = (bf16)v10;
      h1[(16 + mm) * H_ + nt1 * 16 + ncol] = (bf16)v11;
    }
  }
  __syncthreads();

  // ---- layer 2: 32x256 @ 256x256 ----
  v8f c00 = {}, c01 = {}, c10 = {}, c11 = {};
  for (int ks = 0; ks < KS2; ++ks) {
    union { v16bf v; v8bf h[2]; } A0, A1;
    const bf16* ab0 = &h1[m * H_ + ks * 32 + 8 * hh];
    const bf16* ab1 = ab0 + 16 * H_;
    A0.h[0] = *(const v8bf*)(ab0);
    A0.h[1] = *(const v8bf*)(ab0 + 16);
    A1.h[0] = *(const v8bf*)(ab1);
    A1.h[1] = *(const v8bf*)(ab1 + 16);
    v16bf B0 = *(const v16bf*)(w2p + (((ks * 16 + nt0) * 32 + lane) << 4));
    c00 = __builtin_amdgcn_wmma_f32_16x16x32_bf16(false, A0.v, false, B0,
                                                  (short)0, c00, false, false);
    c10 = __builtin_amdgcn_wmma_f32_16x16x32_bf16(false, A1.v, false, B0,
                                                  (short)0, c10, false, false);
    v16bf B1 = *(const v16bf*)(w2p + (((ks * 16 + nt1) * 32 + lane) << 4));
    c01 = __builtin_amdgcn_wmma_f32_16x16x32_bf16(false, A0.v, false, B1,
                                                  (short)0, c01, false, false);
    c11 = __builtin_amdgcn_wmma_f32_16x16x32_bf16(false, A1.v, false, B1,
                                                  (short)0, c11, false, false);
  }

  // ---- bias + relu + layer 3 (dot with W3) directly on D registers ----
  {
    float bb0 = b2[nt0 * 16 + ncol], bb1 = b2[nt1 * 16 + ncol];
    float w30 = W3[nt0 * 16 + ncol], w31 = W3[nt1 * 16 + ncol];
#pragma unroll
    for (int r = 0; r < 8; ++r) {
      float v00 = c00[r] + bb0; v00 = v00 > 0.f ? v00 : 0.f;
      float v01 = c01[r] + bb1; v01 = v01 > 0.f ? v01 : 0.f;
      float v10 = c10[r] + bb0; v10 = v10 > 0.f ? v10 : 0.f;
      float v11 = c11[r] + bb1; v11 = v11 > 0.f ? v11 : 0.f;
      float va = v00 * w30 + v01 * w31;   // sub-tile 0 partial
      float vb = v10 * w30 + v11 * w31;   // sub-tile 1 partial
      // reduce over the 16 lanes that share row m = r + 8*hh
      va += __shfl_xor(va, 1, 16); vb += __shfl_xor(vb, 1, 16);
      va += __shfl_xor(va, 2, 16); vb += __shfl_xor(vb, 2, 16);
      va += __shfl_xor(va, 4, 16); vb += __shfl_xor(vb, 4, 16);
      va += __shfl_xor(va, 8, 16); vb += __shfl_xor(vb, 8, 16);
      if (ncol == 0) {
        atomicAdd(&pl[r + 8 * hh], va);        // ds_add_f32
        atomicAdd(&pl[16 + r + 8 * hh], vb);
      }
    }
  }
  __syncthreads();
  if (tid < MT) {
    predws[((size_t)b * N_ + i) * N_ + j0 + tid] = pl[tid] + b3[0];
  }
}

// ---------------------------------------------------------------------------
// out = 0.5*(p + p^T), zero diagonal
// ---------------------------------------------------------------------------
__global__ void symmetrize(const float* __restrict__ p, float* __restrict__ out) {
  int idx = blockIdx.x * blockDim.x + threadIdx.x;   // < B*N*N
  int j = idx & (N_ - 1);
  int i = (idx >> 9) & (N_ - 1);
  int bb = idx >> 18;
  float v = 0.f;
  if (i != j)
    v = 0.5f * (p[idx] + p[((size_t)bb * N_ + j) * N_ + i]);
  out[idx] = v;
}

extern "C" void kernel_launch(void* const* d_in, const int* in_sizes, int n_in,
                              void* d_out, int out_size, void* d_ws, size_t ws_size,
                              hipStream_t stream) {
  (void)in_sizes; (void)n_in; (void)out_size; (void)ws_size;
  const float* node   = (const float*)d_in[0];
  // d_in[1] = problems (unused; euclid matrix provided)
  const float* euclid = (const float*)d_in[2];
  const float* W1     = (const float*)d_in[3];
  const float* b1     = (const float*)d_in[4];
  const float* W2     = (const float*)d_in[5];
  const float* b2     = (const float*)d_in[6];
  const float* W3     = (const float*)d_in[7];
  const float* b3     = (const float*)d_in[8];

  char* ws = (char*)d_ws;
  bf16*  w1p    = (bf16*)ws;                            // 196608 B
  bf16*  w2p    = (bf16*)(ws + 196608);                 // 131072 B
  float* predws = (float*)(ws + 196608 + 131072);       // 2 MB raw pred

  pack_weights<<<640, 256, 0, stream>>>(W1, W2, w1p, w2p);

  const int tiles = B_ * N_ * (N_ / MT);                // 16384 workgroups
  mlp_edge<<<tiles, 256, 0, stream>>>(node, euclid, w1p, w2p,
                                      W1, b1, b2, W3, b3, predws);

  symmetrize<<<(B_ * N_ * N_) / 256, 256, 0, stream>>>(predws, (float*)d_out);
}
